// CoreAttention_73976516707048
// MI455X (gfx1250) — matrix operand order repaired and checked by reading, not compiled, not run on hardware
//
#include <hip/hip_runtime.h>
#include <stdint.h>

// ---- problem constants (from reference) ----
#define S_LEN 2048
#define BATCH 2
#define HEADS 16
#define DVAL  128
#define BM    128          // Q rows per block (16 per wave, 8 waves)
#define BN    64           // K/V columns per tile
#define NTILE (S_LEN / BN) // 32
#define ROWPITCH (BATCH * HEADS * DVAL) // 4096 floats between consecutive seq rows

typedef __attribute__((ext_vector_type(16))) _Float16 v16h;
typedef __attribute__((ext_vector_type(8)))  _Float16 v8h;
typedef __attribute__((ext_vector_type(4)))  _Float16 v4h;
typedef __attribute__((ext_vector_type(8)))  float    v8f;
typedef unsigned int u32x4 __attribute__((ext_vector_type(4)));
typedef int          i32x8 __attribute__((ext_vector_type(8)));
typedef int          i32x4 __attribute__((ext_vector_type(4)));

static __device__ __forceinline__ v16h cat16(v8h a, v8h b) {
  v16h r;
#pragma unroll
  for (int i = 0; i < 8; ++i) { r[i] = a[i]; r[i + 8] = b[i]; }
  return r;
}

static __device__ __forceinline__ v8f wmma_f16(v16h a, v16h b, v8f c) {
  return __builtin_amdgcn_wmma_f32_16x16x32_f16(
      /*neg_a=*/false, a, /*neg_b=*/false, b,
      /*c_mod=*/(short)0, c, /*reuse_a=*/false, /*reuse_b=*/false);
}

// TDM: DMA one 64x128 f32 tile (row pitch 4096 floats) from global into LDS.
// D# group0: count=1, lds_addr, global_addr[56:0], type=2.
// D# group1: data_size=4B, tensor_dim0=128, tensor_dim1=64, tile_dim0=128,
//            tile_dim1=64, tensor_dim0_stride=4096.
// This toolchain exposes the 6-arg builtin:
//   (uint32x4 g0, int32x8 g1, int32x4 g2, int32x4 g3, int32x8 extra, i32 cpol)
static __device__ __forceinline__ void tdm_load_tile(unsigned lds_addr,
                                                     const float* g) {
  const unsigned long long ga = (unsigned long long)(uintptr_t)g;
  u32x4 g0;
  g0[0] = 1u;                                          // count=1
  g0[1] = lds_addr;                                    // lds_addr (bytes)
  g0[2] = (unsigned)(ga & 0xFFFFFFFFull);              // global_addr[31:0]
  g0[3] = (unsigned)((ga >> 32) & 0x01FFFFFFull)       // global_addr[56:32]
        | (2u << 30);                                  // type=2 ("image")
  i32x8 g1;
  g1[0] = (int)(2u << 16);     // data_size = 2 (4 bytes)
  g1[1] = (int)(128u << 16);   // tensor_dim0 = 128
  g1[2] = (int)(64u  << 16);   // tensor_dim1 = 64
  g1[3] = (int)(128u << 16);   // tile_dim0 = 128
  g1[4] = 64;                  // tile_dim1 = 64
  g1[5] = ROWPITCH;            // tensor_dim0_stride = 4096
  g1[6] = 0;
  g1[7] = 0;
  const i32x4 z4 = {0, 0, 0, 0};
  const i32x8 z8 = {0, 0, 0, 0, 0, 0, 0, 0};
  __builtin_amdgcn_tensor_load_to_lds(g0, g1, z4, z4, z8, 0);
}

__global__ __launch_bounds__(256, 1)
void CoreAttention_fa_kernel(const float* __restrict__ Q,
                             const float* __restrict__ K,
                             const float* __restrict__ V,
                             const uint8_t* __restrict__ MASK,
                             float* __restrict__ OUT) {
  // Double-buffered raw f32 landing zones filled by the Tensor Data Mover,
  // plus f16 operand tiles and a per-wave P pad. 176 KB of 320 KB/WGP.
  __shared__ float    fK[2][BN * DVAL];     // 64 KB
  __shared__ float    fV[2][BN * DVAL];     // 64 KB
  __shared__ _Float16 sK[BN * DVAL];        // 16 KB (key x d, row-major)
  __shared__ _Float16 sVt[DVAL * BN];       // 16 KB (d x key, transposed)
  __shared__ _Float16 sP[8][16 * BN];       // 16 KB

  const int tid  = threadIdx.x;
  const int lane = tid & 31;
  const int wave = tid >> 5;
  const int lanelo = lane & 15;
  const int half   = (lane < 16) ? 0 : 1;   // which 16-lane half
  const int rbase  = half * 8;              // C/D row base for this half

  const int bb = blockIdx.y >> 4;           // batch
  const int hh = blockIdx.y & 15;           // head
  const int qbase = blockIdx.x * BM + wave * 16;
  const int kvhead = (bb * HEADS + hh) * DVAL; // offset of (b,h) within a row

  // ---- load Q A-fragments (16 rows x 128, four K=32 chunks) ----
  // A layout (16-bit, 16x32): lanes 0-15 row=lanelo hold K {0..7,16..23},
  // lanes 16-31 hold K {8..15,24..31}.
  v16h qf[4];
  {
    const int qrow = qbase + lanelo;
    const int base = qrow * ROWPITCH + kvhead;
#pragma unroll
    for (int kc = 0; kc < 4; ++kc) {
      const int koff = kc * 32 + half * 8;
      const float4 a0 = *(const float4*)(Q + base + koff);
      const float4 a1 = *(const float4*)(Q + base + koff + 4);
      const float4 a2 = *(const float4*)(Q + base + koff + 16);
      const float4 a3 = *(const float4*)(Q + base + koff + 20);
      v16h f;
      f[0]=(_Float16)a0.x; f[1]=(_Float16)a0.y; f[2]=(_Float16)a0.z; f[3]=(_Float16)a0.w;
      f[4]=(_Float16)a1.x; f[5]=(_Float16)a1.y; f[6]=(_Float16)a1.z; f[7]=(_Float16)a1.w;
      f[8]=(_Float16)a2.x; f[9]=(_Float16)a2.y; f[10]=(_Float16)a2.z; f[11]=(_Float16)a2.w;
      f[12]=(_Float16)a3.x; f[13]=(_Float16)a3.y; f[14]=(_Float16)a3.z; f[15]=(_Float16)a3.w;
      qf[kc] = f;
    }
  }

  // ---- running softmax state + output accumulators ----
  float m_r[8], l_r[8];
  v8f o[8];
#pragma unroll
  for (int r = 0; r < 8; ++r) { m_r[r] = -INFINITY; l_r[r] = 0.0f; }
#pragma unroll
  for (int n = 0; n < 8; ++n) { v8f z = {}; o[n] = z; }

  const float inv_norm = 0.08838834764831845f; // 1/sqrt(128), coeff=1 cancels
  _Float16* sPw = &sP[wave][0];

  // ---- prologue: kick off DMA for tile 0 ----
  if (wave == 0) {
    const float* kt = K + kvhead;            // tile 0 starts at seq row 0
    const float* vt = V + kvhead;
    tdm_load_tile((unsigned)(uintptr_t)&fK[0][0], kt);
    tdm_load_tile((unsigned)(uintptr_t)&fV[0][0], vt);
  }

  for (int t = 0; t < NTILE; ++t) {
    const int cur = t & 1;
    const int nxt = cur ^ 1;

    if (wave == 0) __builtin_amdgcn_s_wait_tensorcnt(0); // fbuf[cur] landed
    __syncthreads(); // publish fbuf[cur]; all waves done with sK/sVt/fbuf[nxt]

    // ---- async: issue DMA for tile t+1 while we convert + compute tile t ----
    if (wave == 0 && (t + 1) < NTILE) {
      const float* kt = K + (size_t)(t + 1) * BN * ROWPITCH + kvhead;
      const float* vt = V + (size_t)(t + 1) * BN * ROWPITCH + kvhead;
      tdm_load_tile((unsigned)(uintptr_t)&fK[nxt][0], kt);
      tdm_load_tile((unsigned)(uintptr_t)&fV[nxt][0], vt);
    }

    // ---- conversion pass: fbuf[cur] f32 -> sK f16 / sVt f16 (transposed) ----
#pragma unroll
    for (int rep = 0; rep < 8; ++rep) {
      const int idx = rep * 256 + tid;       // 2048 float4 units = 64x128
      const int kl  = idx >> 5;              // local key 0..63
      const int d4  = idx & 31;              // float4 index along d
      const float4 kv = *(const float4*)&fK[cur][kl * DVAL + d4 * 4];
      const float4 vv = *(const float4*)&fV[cur][kl * DVAL + d4 * 4];
      v4h kh; kh[0]=(_Float16)kv.x; kh[1]=(_Float16)kv.y;
              kh[2]=(_Float16)kv.z; kh[3]=(_Float16)kv.w;
      *(v4h*)&sK[kl * DVAL + d4 * 4] = kh;
      sVt[(d4 * 4 + 0) * BN + kl] = (_Float16)vv.x;
      sVt[(d4 * 4 + 1) * BN + kl] = (_Float16)vv.y;
      sVt[(d4 * 4 + 2) * BN + kl] = (_Float16)vv.z;
      sVt[(d4 * 4 + 3) * BN + kl] = (_Float16)vv.w;
    }
    __syncthreads();

    // ---- S = Q K^T for this wave's 16 rows x 64 keys ----
    v8f acc[4];
#pragma unroll
    for (int n = 0; n < 4; ++n) { v8f z = {}; acc[n] = z; }
#pragma unroll
    for (int n = 0; n < 4; ++n) {
      const int key = n * 16 + lanelo;     // B layout: lane -> column (key)
#pragma unroll
      for (int kc = 0; kc < 4; ++kc) {
        const int kb = kc * 32 + half * 8; // lanes 0-15: K {0..7,16..23}; 16-31: {8..15,24..31}
        const v8h b0 = *(const v8h*)&sK[key * DVAL + kb];
        const v8h b1 = *(const v8h*)&sK[key * DVAL + kb + 16];
        acc[n] = wmma_f16(qf[kc], cat16(b0, b1), acc[n]);
      }
    }

    // ---- scale + mask (fp32) ----
#pragma unroll
    for (int n = 0; n < 4; ++n) {
      const int col = t * BN + n * 16 + lanelo;
#pragma unroll
      for (int r = 0; r < 8; ++r) {
        const int row = qbase + rbase + r;
        float v = acc[n][r] * inv_norm;
        const uint8_t mb = MASK[(size_t)(bb * S_LEN + row) * S_LEN + col];
        v = mb ? -10000.0f : v;
        acc[n][r] = v;
      }
    }

    // ---- online softmax update ----
    float alpha[8], mnew[8], rowsum[8];
#pragma unroll
    for (int r = 0; r < 8; ++r) {
      float pm = acc[0][r];
#pragma unroll
      for (int n = 1; n < 4; ++n) pm = fmaxf(pm, acc[n][r]);
#pragma unroll
      for (int sh = 1; sh < 16; sh <<= 1)      // reduce within 16-lane half
        pm = fmaxf(pm, __shfl_xor(pm, sh, 32));
      mnew[r]  = fmaxf(m_r[r], pm);
      alpha[r] = __expf(m_r[r] - mnew[r]);
      m_r[r]   = mnew[r];
      rowsum[r] = 0.0f;
    }

    // P = exp(S - m), write to per-wave LDS pad (row-major 16 x 64, f16)
#pragma unroll
    for (int n = 0; n < 4; ++n) {
#pragma unroll
      for (int r = 0; r < 8; ++r) {
        const float p = __expf(acc[n][r] - mnew[r]);
        rowsum[r] += p;
        sPw[(rbase + r) * BN + n * 16 + lanelo] = (_Float16)p;
      }
    }
#pragma unroll
    for (int r = 0; r < 8; ++r) {
      float s = rowsum[r];
#pragma unroll
      for (int sh = 1; sh < 16; sh <<= 1)
        s += __shfl_xor(s, sh, 32);
      l_r[r] = l_r[r] * alpha[r] + s;
#pragma unroll
      for (int n = 0; n < 8; ++n) o[n][r] *= alpha[r]; // rescale accumulators
    }

    // ---- reload P as A-fragments (two 16x32 chunks); per-wave in-order DS ----
    v16h pf[2];
#pragma unroll
    for (int kc = 0; kc < 2; ++kc) {
      const int off = kc * 32 + half * 8;
      const v8h p0 = *(const v8h*)&sPw[lanelo * BN + off];
      const v8h p1 = *(const v8h*)&sPw[lanelo * BN + off + 16];
      pf[kc] = cat16(p0, p1);
    }

    // ---- O += P V  (8 d-subtiles x 2 K-chunks) ----
#pragma unroll
    for (int n = 0; n < 8; ++n) {
      const int dcol = n * 16 + lanelo;     // B layout: lane -> column (d)
#pragma unroll
      for (int kc = 0; kc < 2; ++kc) {
        const int kb = kc * 32 + half * 8;
        const v8h v0 = *(const v8h*)&sVt[dcol * BN + kb];
        const v8h v1 = *(const v8h*)&sVt[dcol * BN + kb + 16];
        o[n] = wmma_f16(pf[kc], cat16(v0, v1), o[n]);
      }
    }
  }

  // ---- epilogue: normalize and store fp32 (s, b, h*d) ----
#pragma unroll
  for (int r = 0; r < 8; ++r) {
    const int row = qbase + rbase + r;
    const float inv_l = 1.0f / l_r[r];
    const size_t obase = ((size_t)row * BATCH + bb) * (HEADS * DVAL) + hh * DVAL;
#pragma unroll
    for (int n = 0; n < 8; ++n) {
      OUT[obase + n * 16 + lanelo] = o[n][r] * inv_l;
    }
  }
}

extern "C" void kernel_launch(void* const* d_in, const int* in_sizes, int n_in,
                              void* d_out, int out_size, void* d_ws, size_t ws_size,
                              hipStream_t stream) {
  (void)in_sizes; (void)n_in; (void)out_size; (void)d_ws; (void)ws_size;
  const float*   Q    = (const float*)d_in[0];
  const float*   K    = (const float*)d_in[1];
  const float*   V    = (const float*)d_in[2];
  const uint8_t* MASK = (const uint8_t*)d_in[3]; // jnp.bool_ -> 1 byte/elem
  float*         OUT  = (float*)d_out;

  dim3 grid(S_LEN / BM, BATCH * HEADS); // 16 x 32
  dim3 block(256);                      // 8 wave32 waves
  CoreAttention_fa_kernel<<<grid, block, 0, stream>>>(Q, K, V, MASK, OUT);
}
